// NN_29454885716701
// MI455X (gfx1250) — matrix operand order repaired and checked by reading, not compiled, not run on hardware
//
#include <hip/hip_runtime.h>
#include <hip/hip_bf16.h>
#include <stdint.h>

typedef __attribute__((ext_vector_type(16))) _Float16 v16h;
typedef __attribute__((ext_vector_type(8)))  _Float16 v8h;
typedef __attribute__((ext_vector_type(8)))  float    v8f;

#define BETA 0.95f
#define THR  1.0f

constexpr int B_ = 256, T_ = 64, I_ = 2048, H_ = 4096, O_ = 256;
constexpr int M_ = T_ * B_;   // 16384 rows (t*B + b)

// ---------------------------------------------------------------------------
// CDNA5 async global->LDS copy (16B per lane), tracked by ASYNCcnt
// ---------------------------------------------------------------------------
__device__ __forceinline__ void async_copy16(uint32_t lds_byte_off, const void* gptr) {
  asm volatile("global_load_async_to_lds_b128 %0, %1, off"
               :: "v"(lds_byte_off), "v"(gptr) : "memory");
}
__device__ __forceinline__ void wait_async0() {
  asm volatile("s_wait_asynccnt 0x0" ::: "memory");
}

// ---------------------------------------------------------------------------
// X: [B, T, I] f32 (0/1 spikes)  ->  Xh: [T*B, I] f16   (row r = t*B + b)
// ---------------------------------------------------------------------------
__global__ __launch_bounds__(256) void snn_cvt_x(const float* __restrict__ x,
                                                 _Float16* __restrict__ xh) {
  long long idx = (long long)blockIdx.x * blockDim.x + threadIdx.x;  // over T*B*I
  int i = (int)(idx % I_);
  long long r = idx / I_;
  int b = (int)(r % B_);
  int t = (int)(r / B_);
  xh[idx] = (_Float16)x[((long long)b * T_ + t) * I_ + i];
}

// ---------------------------------------------------------------------------
// generic f32 -> f16
// ---------------------------------------------------------------------------
__global__ __launch_bounds__(256) void snn_cvt_w(const float* __restrict__ src,
                                                 _Float16* __restrict__ dst,
                                                 long long n) {
  long long idx = (long long)blockIdx.x * blockDim.x + threadIdx.x;
  if (idx < n) dst[idx] = (_Float16)src[idx];
}

// ---------------------------------------------------------------------------
// C[M,N] = A[M,K] @ Bw[N,K]^T + bias[N]   (A,Bw f16 row-major K-contiguous)
// Block tile 128x128, K-step 64, double-buffered LDS fed by async copies.
// 256 threads = 8 waves in 2(M) x 4(N); each wave: 64x32 = 4x2 WMMA frags.
// ---------------------------------------------------------------------------
__global__ __launch_bounds__(256) void snn_gemm_wmma(
    const _Float16* __restrict__ A, const _Float16* __restrict__ Bw,
    const float* __restrict__ bias, float* __restrict__ C,
    int Ndim, int Kdim) {
  __shared__ _Float16 As[2][128][72];   // +8 halfs pad (16B) vs bank conflicts
  __shared__ _Float16 Bs[2][128][72];

  const int tid   = threadIdx.x;
  const int lane  = tid & 31;
  const int w     = tid >> 5;     // 0..7
  const int wm    = w >> 2;       // 0..1  (64-row slab)
  const int wn    = w & 3;        // 0..3  (32-col slab)
  const int m0    = blockIdx.y * 128;
  const int n0    = blockIdx.x * 128;
  const int lmask = lane & 15;
  const int lhalf = lane >> 4;         // 0 | 1
  const int koffA = lhalf * 8;         // ISA 16-bit A 16x32 layout
  const int koffB = lhalf * 8;         // same k order as A

  // this thread's 4 staged 16B chunks per tile: (row, col) within 128x64 tile
  const int ldr0 = tid >> 3;           // rows tid/8 + {0,32,64,96}
  const int ldc  = (tid & 7) * 8;      // col group

  v8f acc[4][2];
#pragma unroll
  for (int i = 0; i < 4; ++i)
#pragma unroll
    for (int j = 0; j < 2; ++j) {
      v8f z = {0.f, 0.f, 0.f, 0.f, 0.f, 0.f, 0.f, 0.f};
      acc[i][j] = z;
    }

  const int ntiles = Kdim / 64;

  // issue async copies for one 128x64 A-tile + B-tile into buffer p
  auto issue_tile = [&](int tile, int p) {
#pragma unroll
    for (int l = 0; l < 4; ++l) {
      int r = ldr0 + l * 32;
      int kcol = tile * 64 + ldc;
      async_copy16((uint32_t)(uintptr_t)&As[p][r][ldc],
                   &A[(long long)(m0 + r) * Kdim + kcol]);
      async_copy16((uint32_t)(uintptr_t)&Bs[p][r][ldc],
                   &Bw[(long long)(n0 + r) * Kdim + kcol]);
    }
  };

  issue_tile(0, 0);

  for (int it = 0; it < ntiles; ++it) {
    const int p = it & 1;
    wait_async0();        // this wave's copies for tile `it` landed in LDS
    __syncthreads();      // all waves: tile `it` ready AND buffer 1-p free
    if (it + 1 < ntiles) issue_tile(it + 1, p ^ 1);

#pragma unroll
    for (int kk = 0; kk < 64; kk += 32) {
      v16h afrag[4], bfrag[2];
#pragma unroll
      for (int i = 0; i < 4; ++i) {
        const _Float16* pa = &As[p][wm * 64 + i * 16 + lmask][kk + koffA];
#pragma unroll
        for (int j = 0; j < 8; ++j) { afrag[i][j] = pa[j]; afrag[i][8 + j] = pa[16 + j]; }
      }
#pragma unroll
      for (int i = 0; i < 2; ++i) {
        const _Float16* pb = &Bs[p][wn * 32 + i * 16 + lmask][kk + koffB];
#pragma unroll
        for (int j = 0; j < 8; ++j) { bfrag[i][j] = pb[j]; bfrag[i][8 + j] = pb[16 + j]; }
      }
#pragma unroll
      for (int i = 0; i < 4; ++i)
#pragma unroll
        for (int j = 0; j < 2; ++j)
          acc[i][j] = __builtin_amdgcn_wmma_f32_16x16x32_f16(
              false, afrag[i], false, bfrag[j], (short)0, acc[i][j], false, false);
    }
  }

  // epilogue: C layout — VGPR r: lanes0-15 M=r, lanes16-31 M=8+r; N=lane&15
#pragma unroll
  for (int i = 0; i < 4; ++i) {
    int row = m0 + wm * 64 + i * 16 + lhalf * 8;
#pragma unroll
    for (int j = 0; j < 2; ++j) {
      int col = n0 + wn * 32 + j * 16 + lmask;
      float bv = bias[col];
#pragma unroll
      for (int r = 0; r < 8; ++r)
        C[(long long)(row + r) * Ndim + col] = acc[i][j][r] + bv;
    }
  }
}

// ---------------------------------------------------------------------------
// hidden-layer LIF recurrence: per (b,h) over t; spikes written as f16 0/1
// (ready to be the A matrix of the layer-2 WMMA GEMM)
// ---------------------------------------------------------------------------
__global__ __launch_bounds__(256) void snn_rec_hidden(const float* __restrict__ cur,
                                                      _Float16* __restrict__ spkh) {
  long long idx = (long long)blockIdx.x * blockDim.x + threadIdx.x;  // over B*H
  const long long stride = (long long)B_ * H_;
  float mem = 0.f;
#pragma unroll 1
  for (int t = 0; t < T_; ++t) {
    float reset = (mem > THR) ? THR : 0.f;          // from PREVIOUS mem (detached)
    mem = BETA * mem + cur[(long long)t * stride + idx] - reset;
    spkh[(long long)t * stride + idx] = (mem > THR) ? (_Float16)1.0f : (_Float16)0.0f;
  }
}

// ---------------------------------------------------------------------------
// output-layer LIF recurrence: writes [spk_trace | mem_trace] (both [T,B,O] f32)
// ---------------------------------------------------------------------------
__global__ __launch_bounds__(256) void snn_rec_out(const float* __restrict__ cur,
                                                   float* __restrict__ out) {
  int idx = blockIdx.x * blockDim.x + threadIdx.x;   // over B*O
  const int stride = B_ * O_;
  const int tbo = T_ * stride;
  float mem = 0.f;
#pragma unroll 1
  for (int t = 0; t < T_; ++t) {
    float reset = (mem > THR) ? THR : 0.f;
    mem = BETA * mem + cur[t * stride + idx] - reset;
    out[t * stride + idx] = (mem > THR) ? 1.f : 0.f;  // spk_trace
    out[tbo + t * stride + idx] = mem;                // mem_trace
  }
}

// ---------------------------------------------------------------------------
extern "C" void kernel_launch(void* const* d_in, const int* in_sizes, int n_in,
                              void* d_out, int out_size, void* d_ws, size_t ws_size,
                              hipStream_t stream) {
  const float* x  = (const float*)d_in[0];   // [B,T,I]
  const float* W1 = (const float*)d_in[1];   // [H,I]
  const float* b1 = (const float*)d_in[2];   // [H]
  const float* W2 = (const float*)d_in[3];   // [O,H]
  const float* b2 = (const float*)d_in[4];   // [O]
  float* out = (float*)d_out;

  // workspace carve-up
  char* ws = (char*)d_ws;
  _Float16* Xh  = (_Float16*)ws;                               // M*I
  _Float16* W1h = Xh  + (size_t)M_ * I_;                       // H*I
  _Float16* W2h = W1h + (size_t)H_ * I_;                       // O*H
  float*    cur1 = (float*)(W2h + (size_t)O_ * H_);            // M*H
  float*    cur2 = cur1 + (size_t)M_ * H_;                     // M*O
  _Float16* spkh = (_Float16*)(cur2 + (size_t)M_ * O_);        // M*H

  // 1) precision conversion / layout
  snn_cvt_x<<<(int)(((long long)M_ * I_) / 256), 256, 0, stream>>>(x, Xh);
  snn_cvt_w<<<(int)(((long long)H_ * I_ + 255) / 256), 256, 0, stream>>>(W1, W1h, (long long)H_ * I_);
  snn_cvt_w<<<(int)(((long long)O_ * H_ + 255) / 256), 256, 0, stream>>>(W2, W2h, (long long)O_ * H_);

  // 2) layer-1 batched-over-time GEMM: cur1[M,H] = Xh @ W1h^T + b1
  snn_gemm_wmma<<<dim3(H_ / 128, M_ / 128), 256, 0, stream>>>(Xh, W1h, b1, cur1, H_, I_);

  // 3) hidden LIF recurrence -> spk_h (f16 0/1)
  snn_rec_hidden<<<(int)(((long long)B_ * H_) / 256), 256, 0, stream>>>(cur1, spkh);

  // 4) layer-2 GEMM: cur2[M,O] = spkh @ W2h^T + b2
  snn_gemm_wmma<<<dim3(O_ / 128, M_ / 128), 256, 0, stream>>>(spkh, W2h, b2, cur2, O_, H_);

  // 5) output LIF recurrence -> [spk_trace | mem_trace]
  snn_rec_out<<<(B_ * O_) / 256, 256, 0, stream>>>(cur2, out);
}
